// TransformerBlock_70300024701812
// MI455X (gfx1250) — matrix operand order repaired and checked by reading, not compiled, not run on hardware
//
#include <hip/hip_runtime.h>

// ---------------------------------------------------------------------------
// Transformer block for MI455X (gfx1250): bf16 WMMA GEMMs (f32 accumulate),
// 320KB-LDS-resident attention, fused bias/ReLU/residual/LN epilogues.
// ---------------------------------------------------------------------------

#define DEV __device__ __forceinline__

typedef __bf16 bf16;
typedef __attribute__((ext_vector_type(16))) __bf16 v16bf;
typedef __attribute__((ext_vector_type(8)))  __bf16 v8bf;
typedef __attribute__((ext_vector_type(8)))  float  v8f;

constexpr int Bn  = 8;
constexpr int Sn  = 1024;
constexpr int Hn  = 1024;
constexpr int NHn = 16;
constexpr int DHn = 64;
constexpr int Mn  = Bn * Sn;          // 8192 token rows

DEV v8f zero8() {
  v8f z = {0.f, 0.f, 0.f, 0.f, 0.f, 0.f, 0.f, 0.f};
  return z;
}

DEV v8f wmma_bf16(v16bf a, v16bf b, v8f c) {
  // D = A x B + C ; f32 accumulate, 16x16x32 bf16 (8-arg form)
  return __builtin_amdgcn_wmma_f32_16x16x32_bf16(
      /*neg_a=*/false, a, /*neg_b=*/false, b,
      /*c_mod=*/(short)0, c, /*reuse_a=*/false, /*reuse_b=*/false);
}

DEV v16bf combine8(v8bf lo, v8bf hi) {
  v16bf r;
#pragma unroll
  for (int e = 0; e < 8; ++e) { r[e] = lo[e]; r[8 + e] = hi[e]; }
  return r;
}

// ---------------------------------------------------------------------------
// Weight prep: Wt[n][k] = bf16(W[k][n])   (transpose + downconvert, one-time)
// ---------------------------------------------------------------------------
__global__ void __launch_bounds__(256)
k_prep_wt(const float* __restrict__ w, bf16* __restrict__ wt, int N, int K) {
  int idx = blockIdx.x * 256 + threadIdx.x;
  if (idx >= N * K) return;
  int n = idx / K, k = idx % K;
  wt[idx] = (bf16)w[(size_t)k * N + n];
}

// ---------------------------------------------------------------------------
// Row LayerNorm over H=1024. Block=256 threads (8 waves), 4 cols/thread.
// Optional f32 out, optional bf16 out, optional post-LN residual add.
// ---------------------------------------------------------------------------
__global__ void __launch_bounds__(256)
k_layernorm(const float* __restrict__ in, const float* __restrict__ g,
            const float* __restrict__ be, const float* __restrict__ post,
            float* __restrict__ outf, bf16* __restrict__ outbf) {
  const int row = blockIdx.x;
  const int tid = threadIdx.x;
  const float* __restrict__ rp = in + (size_t)row * Hn;

  float v[4];
  float s = 0.f, q = 0.f;
#pragma unroll
  for (int p = 0; p < 4; ++p) {
    v[p] = rp[tid + p * 256];
    s += v[p];
    q += v[p] * v[p];
  }
#pragma unroll
  for (int off = 16; off; off >>= 1) {      // wave32 reduce
    s += __shfl_xor(s, off, 32);
    q += __shfl_xor(q, off, 32);
  }
  __shared__ float rs[8], rq[8];
  __shared__ float meanS, rstdS;
  const int wid = tid >> 5, lane = tid & 31;
  if (lane == 0) { rs[wid] = s; rq[wid] = q; }
  __syncthreads();
  if (tid == 0) {
    float ts = 0.f, tq = 0.f;
    for (int i = 0; i < 8; ++i) { ts += rs[i]; tq += rq[i]; }
    float mean = ts * (1.0f / Hn);
    float var  = tq * (1.0f / Hn) - mean * mean;
    meanS = mean;
    rstdS = rsqrtf(var + 1e-5f);
  }
  __syncthreads();
  const float mean = meanS, rstd = rstdS;
#pragma unroll
  for (int p = 0; p < 4; ++p) {
    const int col = tid + p * 256;
    float val = (v[p] - mean) * rstd * g[col] + be[col];
    if (outbf) outbf[(size_t)row * Hn + col] = (bf16)val;
    if (outf) {
      float o = val;
      if (post) o += post[(size_t)row * Hn + col];
      outf[(size_t)row * Hn + col] = o;
    }
  }
}

// ---------------------------------------------------------------------------
// GEMM: C[M,N] = A[M,K](bf16) * Wt[N,K](bf16, pre-transposed) + bias
// Block = 256 threads = 8 waves (2x4), block tile 128x256, wave tile 64x64.
// mode 0: plain [M,N] f32 and/or bf16 out (+relu, +residual)
// mode 1: bf16 out scattered to [b, head, s, d]   (Q / K head layout)
// mode 2: bf16 out scattered to [b, head, d, s]   (V transposed layout)
// ---------------------------------------------------------------------------
__global__ void __launch_bounds__(256)
k_gemm(const bf16* __restrict__ A, const bf16* __restrict__ Wt,
       const float* __restrict__ bias, const float* __restrict__ resid,
       float* __restrict__ outf, bf16* __restrict__ outbf,
       int Kd, int Nd, int relu, int mode) {
  const int tid  = threadIdx.x;
  const int wid  = tid >> 5;
  const int lane = tid & 31;
  const int ml   = lane & 15;       // M (or N) index within 16-lane group
  const int hf   = lane >> 4;       // lane-half selects K sub-range
  const int m0 = blockIdx.x * 128 + (wid & 1) * 64;
  const int n0 = blockIdx.y * 256 + (wid >> 1) * 64;
  const int kh8  = hf * 8;          // A prefetch only: K offset 0 / 8 per half
  const int kb16 = hf * 16;         // B frag: K 0..15 / 16..31 per half

  v8f c[4][4];
#pragma unroll
  for (int i = 0; i < 4; ++i)
#pragma unroll
    for (int j = 0; j < 4; ++j) c[i][j] = zero8();

  for (int k = 0; k < Kd; k += 32) {
    // Unconditional speculative prefetch of next K-step (one per side; a
    // translation miss past the end is silently dropped -> safe).
    __builtin_prefetch(A  + (size_t)(m0 + ml) * Kd + k + kh8  + 32, 0, 1);
    __builtin_prefetch(Wt + (size_t)(n0 + ml) * Kd + k + kb16 + 32, 0, 1);

    v16bf a[4], b[4];
#pragma unroll
    for (int i = 0; i < 4; ++i) {
      const bf16* p = A + (size_t)(m0 + 16 * i + ml) * Kd + k + kb16;
      a[i] = combine8(*(const v8bf*)p, *(const v8bf*)(p + 8));
    }
#pragma unroll
    for (int j = 0; j < 4; ++j)
      b[j] = *(const v16bf*)(Wt + (size_t)(n0 + 16 * j + ml) * Kd + k + kb16);
#pragma unroll
    for (int i = 0; i < 4; ++i)
#pragma unroll
      for (int j = 0; j < 4; ++j)
        c[i][j] = wmma_bf16(a[i], b[j], c[i][j]);
  }

  // Epilogue. C/D layout: VGPR r -> row 16i + r + 8*hf ; col = 16j + ml.
#pragma unroll
  for (int i = 0; i < 4; ++i) {
#pragma unroll
    for (int j = 0; j < 4; ++j) {
#pragma unroll
      for (int r = 0; r < 8; ++r) {
        const int row = m0 + 16 * i + r + 8 * hf;
        const int col = n0 + 16 * j + ml;
        float val = c[i][j][r] + bias[col];
        if (relu) val = fmaxf(val, 0.f);
        if (resid) val += resid[(size_t)row * Nd + col];
        if (mode == 0) {
          if (outf)  outf [(size_t)row * Nd + col] = val;
          if (outbf) outbf[(size_t)row * Nd + col] = (bf16)val;
        } else {
          const int bb = row >> 10, ss = row & (Sn - 1);
          const int hd = col >> 6,  dd = col & (DHn - 1);
          size_t idx;
          if (mode == 1) idx = ((size_t)(bb * NHn + hd) * Sn + ss) * DHn + dd;
          else           idx = ((size_t)(bb * NHn + hd) * DHn + dd) * Sn + ss;
          outbf[idx] = (bf16)val;
        }
      }
    }
  }
}

// ---------------------------------------------------------------------------
// Attention: one WG per (q-slab of 64, head, batch). Scores 64x1024 f32 stay
// resident in LDS (256 KB of the 320 KB WGP LDS). 256 threads = 8 waves.
//   phase 1: S = (Q K^T) * scale  via WMMA, each wave owns 128 key columns
//   phase 2: in-LDS row softmax (mask per query row)
//   phase 3: O = P V via WMMA (K split over waves), ds_add_f32 combine
//   phase 4: scatter O to [b, s, h*DH + d] in bf16 for the output projection
// ---------------------------------------------------------------------------
constexpr size_t ATTN_LDS = (size_t)64 * 1024 * 4 + (size_t)64 * 64 * 4;

__global__ void __launch_bounds__(256)
k_attn(const bf16* __restrict__ Qh, const bf16* __restrict__ Kh,
       const bf16* __restrict__ Vth, const int* __restrict__ mask,
       bf16* __restrict__ out_bf, float scale) {
  extern __shared__ char smem_raw[];
  float* scores = (float*)smem_raw;                               // [64][1024]
  float* oacc   = (float*)(smem_raw + (size_t)64 * 1024 * 4);     // [64][64]

  const int b  = blockIdx.z;
  const int hh = blockIdx.y;
  const int q0 = blockIdx.x * 64;
  const int tid  = threadIdx.x;
  const int wid  = tid >> 5;
  const int lane = tid & 31;
  const int ml = lane & 15, hf = lane >> 4;

  const bf16* Q  = Qh  + ((size_t)(b * NHn + hh) * Sn) * DHn;
  const bf16* Kb = Kh  + ((size_t)(b * NHn + hh) * Sn) * DHn;
  const bf16* Vt = Vth + ((size_t)(b * NHn + hh) * DHn) * Sn;

  const int key0 = wid * 128;   // this wave's 128 key columns

  // ---- phase 1: scores ----
  for (int jo = 0; jo < 2; ++jo) {
    const int keyb = key0 + jo * 64;
    v8f c[4][4];
#pragma unroll
    for (int i = 0; i < 4; ++i)
#pragma unroll
      for (int j = 0; j < 4; ++j) c[i][j] = zero8();

#pragma unroll
    for (int kk = 0; kk < DHn; kk += 32) {
      v16bf a[4], bm[4];
#pragma unroll
      for (int i = 0; i < 4; ++i) {
        const bf16* p = Q + (size_t)(q0 + 16 * i + ml) * DHn + kk + hf * 16;
        a[i] = combine8(*(const v8bf*)p, *(const v8bf*)(p + 8));
      }
#pragma unroll
      for (int j = 0; j < 4; ++j)
        bm[j] = *(const v16bf*)(Kb + (size_t)(keyb + 16 * j + ml) * DHn + kk + hf * 16);
#pragma unroll
      for (int i = 0; i < 4; ++i)
#pragma unroll
        for (int j = 0; j < 4; ++j)
          c[i][j] = wmma_bf16(a[i], bm[j], c[i][j]);
    }
#pragma unroll
    for (int i = 0; i < 4; ++i)
#pragma unroll
      for (int j = 0; j < 4; ++j)
#pragma unroll
        for (int r = 0; r < 8; ++r)
          scores[(size_t)(16 * i + r + 8 * hf) * 1024 + keyb + 16 * j + ml] =
              c[i][j][r] * scale;
  }
  __syncthreads();

  // ---- phase 2: softmax (8 rows per wave) ----
  for (int rr = 0; rr < 8; ++rr) {
    const int row = wid * 8 + rr;
    float* srow = scores + (size_t)row * 1024;
    const int mval = mask[b * Sn + q0 + row];
    if (mval == 0) {
      for (int c0 = lane; c0 < 1024; c0 += 32) srow[c0] = 1.0f / 1024.0f;
    } else {
      float mx = -1e30f;
      for (int c0 = lane; c0 < 1024; c0 += 32) mx = fmaxf(mx, srow[c0]);
#pragma unroll
      for (int off = 16; off; off >>= 1) mx = fmaxf(mx, __shfl_xor(mx, off, 32));
      float sum = 0.f;
      for (int c0 = lane; c0 < 1024; c0 += 32) {
        float e = __expf(srow[c0] - mx);
        srow[c0] = e;
        sum += e;
      }
#pragma unroll
      for (int off = 16; off; off >>= 1) sum += __shfl_xor(sum, off, 32);
      const float inv = 1.0f / sum;
      for (int c0 = lane; c0 < 1024; c0 += 32) srow[c0] *= inv;
    }
  }
  __syncthreads();

  for (int idx = tid; idx < 64 * 64; idx += 256) oacc[idx] = 0.f;
  __syncthreads();

  // ---- phase 3: O = P V (wave covers keys [key0, key0+128)) ----
  {
    v8f c[4][4];
#pragma unroll
    for (int i = 0; i < 4; ++i)
#pragma unroll
      for (int j = 0; j < 4; ++j) c[i][j] = zero8();

#pragma unroll
    for (int ks = 0; ks < 128; ks += 32) {
      const int kg = key0 + ks;
      v16bf a[4], bm[4];
#pragma unroll
      for (int i = 0; i < 4; ++i) {
        const float* sp = scores + (size_t)(16 * i + ml) * 1024 + kg + hf * 16;
        v16bf af;
#pragma unroll
        for (int e = 0; e < 8; ++e) {
          af[e]     = (bf16)sp[e];
          af[8 + e] = (bf16)sp[e + 8];
        }
        a[i] = af;
      }
#pragma unroll
      for (int j = 0; j < 4; ++j)
        bm[j] = *(const v16bf*)(Vt + (size_t)(16 * j + ml) * Sn + kg + hf * 16);
#pragma unroll
      for (int i = 0; i < 4; ++i)
#pragma unroll
        for (int j = 0; j < 4; ++j)
          c[i][j] = wmma_bf16(a[i], bm[j], c[i][j]);
    }
#pragma unroll
    for (int i = 0; i < 4; ++i)
#pragma unroll
      for (int j = 0; j < 4; ++j)
#pragma unroll
        for (int r = 0; r < 8; ++r)
          atomicAdd(&oacc[(16 * i + r + 8 * hf) * 64 + 16 * j + ml], c[i][j][r]);
  }
  __syncthreads();

  // ---- phase 4: scatter to [b, s, hh*DH + d] ----
  for (int idx = tid; idx < 64 * 64; idx += 256) {
    const int row = idx >> 6, col = idx & 63;
    out_bf[((size_t)b * Sn + q0 + row) * Hn + hh * DHn + col] = (bf16)oacc[idx];
  }
}

// ---------------------------------------------------------------------------
// Orchestration
// ---------------------------------------------------------------------------
extern "C" void kernel_launch(void* const* d_in, const int* in_sizes, int n_in,
                              void* d_out, int out_size, void* d_ws, size_t ws_size,
                              hipStream_t stream) {
  (void)in_sizes; (void)n_in; (void)out_size; (void)ws_size;

  const float* x    = (const float*)d_in[0];
  const float* h    = (const float*)d_in[1];
  const int*   mask = (const int*)d_in[2];
  const float* W[10];
  const float* Bi[10];
  for (int i = 0; i < 10; ++i) W[i]  = (const float*)d_in[3 + i];
  for (int i = 0; i < 10; ++i) Bi[i] = (const float*)d_in[13 + i];
  const float* ln1_g  = (const float*)d_in[23], *ln1_b  = (const float*)d_in[24];
  const float* ln2_g  = (const float*)d_in[25], *ln2_b  = (const float*)d_in[26];
  const float* ln3_g  = (const float*)d_in[27], *ln3_b  = (const float*)d_in[28];
  const float* lnh_g  = (const float*)d_in[29], *lnh_b  = (const float*)d_in[30];
  const float* ffln_g = (const float*)d_in[31], *ffln_b = (const float*)d_in[32];
  float* out = (float*)d_out;

  // workspace carve-up (all offsets 256B aligned)
  char* ws = (char*)d_ws;
  size_t off = 0;
  auto carve = [&](size_t bytes) {
    char* p = ws + off;
    off += (bytes + 255) & ~(size_t)255;
    return (void*)p;
  };
  bf16* wt[10];
  for (int i = 0; i < 10; ++i) wt[i] = (bf16*)carve((size_t)Hn * Hn * 2);
  bf16* bf0 = (bf16*)carve((size_t)Mn * Hn * 2);  // LN outputs feeding GEMMs
  bf16* bf1 = (bf16*)carve((size_t)Mn * Hn * 2);  // hn / ff1
  bf16* qb  = (bf16*)carve((size_t)Mn * Hn * 2);  // Q [b,h,s,d]
  bf16* kb  = (bf16*)carve((size_t)Mn * Hn * 2);  // K [b,h,s,d]
  bf16* vtb = (bf16*)carve((size_t)Mn * Hn * 2);  // V^T [b,h,d,s]
  bf16* ab  = (bf16*)carve((size_t)Mn * Hn * 2);  // attention out (bf16)
  float* f0 = (float*)carve((size_t)Mn * Hn * 4); // hidden_in
  float* f1 = (float*)carve((size_t)Mn * Hn * 4); // hidden_mid
  float* f2 = (float*)carve((size_t)Mn * Hn * 4); // y, then ff+y

  hipFuncSetAttribute((const void*)k_attn,
                      hipFuncAttributeMaxDynamicSharedMemorySize, (int)ATTN_LDS);

  const dim3 gGemm(Mn / 128, Hn / 256);
  const dim3 gAttn(Sn / 64, NHn, Bn);
  const int TPB = 256;
  const int wtBlocks = (Hn * Hn + TPB - 1) / TPB;

  for (int i = 0; i < 10; ++i)
    k_prep_wt<<<wtBlocks, TPB, 0, stream>>>(W[i], wt[i], Hn, Hn);

  // ---- self-attention block ----
  k_layernorm<<<Mn, TPB, 0, stream>>>(x, ln1_g, ln1_b, nullptr, nullptr, bf0);
  k_gemm<<<gGemm, TPB, 0, stream>>>(bf0, wt[0], Bi[0], nullptr, nullptr, qb,  Hn, Hn, 0, 1);
  k_gemm<<<gGemm, TPB, 0, stream>>>(bf0, wt[1], Bi[1], nullptr, nullptr, kb,  Hn, Hn, 0, 1);
  k_gemm<<<gGemm, TPB, 0, stream>>>(bf0, wt[2], Bi[2], nullptr, nullptr, vtb, Hn, Hn, 0, 2);
  k_attn<<<gAttn, TPB, ATTN_LDS, stream>>>(qb, kb, vtb, mask, ab, 0.125f);
  k_gemm<<<gGemm, TPB, 0, stream>>>(ab, wt[3], Bi[3], x, f0, nullptr, Hn, Hn, 0, 0); // + x

  // ---- cross-attention block ----
  k_layernorm<<<Mn, TPB, 0, stream>>>(f0, ln2_g, ln2_b, nullptr, nullptr, bf0);
  k_layernorm<<<Mn, TPB, 0, stream>>>(h,  lnh_g, lnh_b, nullptr, nullptr, bf1);
  k_gemm<<<gGemm, TPB, 0, stream>>>(bf0, wt[4], Bi[4], nullptr, nullptr, qb,  Hn, Hn, 0, 1);
  k_gemm<<<gGemm, TPB, 0, stream>>>(bf1, wt[5], Bi[5], nullptr, nullptr, kb,  Hn, Hn, 0, 1);
  k_gemm<<<gGemm, TPB, 0, stream>>>(bf1, wt[6], Bi[6], nullptr, nullptr, vtb, Hn, Hn, 0, 2);
  k_attn<<<gAttn, TPB, ATTN_LDS, stream>>>(qb, kb, vtb, mask, ab, 0.125f);
  k_gemm<<<gGemm, TPB, 0, stream>>>(ab, wt[7], Bi[7], f0, f1, nullptr, Hn, Hn, 0, 0); // + hidden_in

  // ---- FFN block ----
  k_layernorm<<<Mn, TPB, 0, stream>>>(f1, ln3_g, ln3_b, nullptr, f2, bf0);           // y (f32+bf16)
  k_gemm<<<gGemm, TPB, 0, stream>>>(bf0, wt[8], Bi[8], nullptr, nullptr, bf1, Hn, Hn, 1, 0); // relu
  k_gemm<<<gGemm, TPB, 0, stream>>>(bf1, wt[9], Bi[9], f2, f2, nullptr, Hn, Hn, 0, 0);       // ff + y
  k_layernorm<<<Mn, TPB, 0, stream>>>(f2, ffln_g, ffln_b, f1, out, nullptr);         // LN + hidden_mid
}
